// GatedDeltaNet_53068615909628
// MI455X (gfx1250) — compile-verified, matrix operand order checked by reading, not run on hardware
//
#include <hip/hip_runtime.h>
#include <hip/hip_bf16.h>

typedef _Float16 v16h __attribute__((ext_vector_type(16)));
typedef _Float16 v8h  __attribute__((ext_vector_type(8)));
typedef float    v8f  __attribute__((ext_vector_type(8)));

constexpr int Bb = 2, Tt = 2048, Dm = 2048, H = 16, DK = 128, DV = 128, KW = 4;
constexpr int CONV = 2 * H * DK + H * DV;   // 6144
constexpr int BT = Bb * Tt;                 // 4096
constexpr float EPSF = 1e-6f;

// Low 32 bits of a flat shared-aperture pointer == wave-relative LDS byte
// offset (ISA 10.2); this is what the async
// to-LDS instructions expect in VDST.
__device__ __forceinline__ unsigned ldsoff(const void* p) {
    return (unsigned)(unsigned long long)p;
}

__device__ __forceinline__ void async_load_b128(unsigned lds_dst, const void* gsrc) {
    asm volatile("global_load_async_to_lds_b128 %0, %1, off"
                 :: "v"(lds_dst), "v"(gsrc) : "memory");
}
__device__ __forceinline__ void async_load_b32(unsigned lds_dst, const void* gsrc) {
    asm volatile("global_load_async_to_lds_b32 %0, %1, off"
                 :: "v"(lds_dst), "v"(gsrc) : "memory");
}
__device__ __forceinline__ void wait_async0() {
    asm volatile("s_wait_asynccnt 0x0" ::: "memory");
}

// ---------------------------------------------------------------------------
// fp32 -> f16 conversion
// ---------------------------------------------------------------------------
__global__ __launch_bounds__(256) void cvt_f32_f16(const float* __restrict__ in,
                                                   _Float16* __restrict__ out, int n) {
    int i = blockIdx.x * 256 + threadIdx.x;
    if (i < n) out[i] = (_Float16)in[i];
}

// ---------------------------------------------------------------------------
// C[M,N] (f32) = A[M,K] (f16) * B[N,K]^T (f16), fp32 accumulate via WMMA.
// 256 threads / 8 waves -> 128x128 C tile; wave computes 32x64 (8 WMMA/Kstep).
// Double-buffered LDS filled by GLOBAL_LOAD_ASYNC_TO_LDS_B128 (ASYNCcnt),
// overlapping the DMA for tile k+1 with the WMMAs of tile k.
// ---------------------------------------------------------------------------
__global__ __launch_bounds__(256) void gemm_f16_abt(const _Float16* __restrict__ A,
                                                    const _Float16* __restrict__ Bm,
                                                    float* __restrict__ C,
                                                    int M, int N, int K) {
    constexpr int LS = 40; // LDS row stride in halves (80B)
    __shared__ __align__(16) _Float16 As[2][128 * LS];
    __shared__ __align__(16) _Float16 Bs[2][128 * LS];

    const int tid  = threadIdx.x;
    const int lane = tid & 31;
    const int wv   = tid >> 5;
    const int mrow = (wv & 3) * 32;     // wave M offset in tile
    const int ncol = (wv >> 2) * 64;    // wave N offset in tile
    const int r    = lane & 15;
    const int koff = (lane >> 4) * 8;
    const int mBase = blockIdx.y * 128;
    const int nBase = blockIdx.x * 128;

    // Loader mapping: tile is 128 rows x 32 halves = 512 b128-chunks per
    // matrix; each thread owns 2 chunks per matrix (4 async loads / K-step).
    const int ar0 = tid >> 2,         ac0 = (tid & 3) * 8;
    const int ar1 = (tid + 256) >> 2, ac1 = ((tid + 256) & 3) * 8;
    const _Float16* gA0 = A  + (size_t)(mBase + ar0) * K + ac0;
    const _Float16* gA1 = A  + (size_t)(mBase + ar1) * K + ac1;
    const _Float16* gB0 = Bm + (size_t)(nBase + ar0) * K + ac0;
    const _Float16* gB1 = Bm + (size_t)(nBase + ar1) * K + ac1;
    unsigned lA0[2], lA1[2], lB0[2], lB1[2];
#pragma unroll
    for (int s = 0; s < 2; ++s) {
        lA0[s] = ldsoff(&As[s][ar0 * LS + ac0]);
        lA1[s] = ldsoff(&As[s][ar1 * LS + ac1]);
        lB0[s] = ldsoff(&Bs[s][ar0 * LS + ac0]);
        lB1[s] = ldsoff(&Bs[s][ar1 * LS + ac1]);
    }

    v8f acc[2][4];
#pragma unroll
    for (int i = 0; i < 2; ++i)
#pragma unroll
        for (int j = 0; j < 4; ++j) acc[i][j] = (v8f){};

    // prologue: async-fill buffer 0 with K-tile 0
    async_load_b128(lA0[0], gA0);
    async_load_b128(lA1[0], gA1);
    async_load_b128(lB0[0], gB0);
    async_load_b128(lB1[0], gB1);

    for (int k0 = 0; k0 < K; k0 += 32) {
        const int s = (k0 >> 5) & 1;
        wait_async0();       // this wave's tile-k data is in LDS
        __syncthreads();     // all waves' data in LDS; prev compute done
        if (k0 + 32 < K) {   // kick off DMA for tile k+1 into other buffer
            const int sn = s ^ 1;
            async_load_b128(lA0[sn], gA0 + k0 + 32);
            async_load_b128(lA1[sn], gA1 + k0 + 32);
            async_load_b128(lB0[sn], gB0 + k0 + 32);
            async_load_b128(lB1[sn], gB1 + k0 + 32);
        }

        v16h a[2], b[4];
#pragma unroll
        for (int i = 0; i < 2; ++i) {
            const _Float16* ap = &As[s][(mrow + i * 16 + r) * LS + koff];
            v8h lo = *(const v8h*)ap, hi = *(const v8h*)(ap + 16);
            a[i] = __builtin_shufflevector(lo, hi, 0,1,2,3,4,5,6,7,8,9,10,11,12,13,14,15);
        }
#pragma unroll
        for (int j = 0; j < 4; ++j) {
            const _Float16* bp = &Bs[s][(ncol + j * 16 + r) * LS + koff];
            v8h lo = *(const v8h*)bp, hi = *(const v8h*)(bp + 16);
            b[j] = __builtin_shufflevector(lo, hi, 0,1,2,3,4,5,6,7,8,9,10,11,12,13,14,15);
        }
#pragma unroll
        for (int i = 0; i < 2; ++i)
#pragma unroll
            for (int j = 0; j < 4; ++j)
                acc[i][j] = __builtin_amdgcn_wmma_f32_16x16x32_f16(
                    false, a[i], false, b[j], (short)0, acc[i][j], false, false);
    }

#pragma unroll
    for (int i = 0; i < 2; ++i) {
#pragma unroll
        for (int j = 0; j < 4; ++j) {
            const int cm = mBase + mrow + i * 16 + (lane >> 4) * 8;
            const int cn = nBase + ncol + j * 16 + r;
#pragma unroll
            for (int q = 0; q < 8; ++q)
                C[(size_t)(cm + q) * N + cn] = acc[i][j][q];
        }
    }
}

// ---------------------------------------------------------------------------
// Causal depthwise conv (KW=4) + SiLU + per-head l2norm of q,k; copy v.
// ---------------------------------------------------------------------------
__global__ __launch_bounds__(256) void conv_silu_norm(const float* __restrict__ mixed,
                                                      const float* __restrict__ cw,
                                                      float* __restrict__ qn,
                                                      float* __restrict__ kn,
                                                      float* __restrict__ vbuf) {
    __shared__ float yrow[CONV];
    const int bt = blockIdx.x;
    const int t  = bt % Tt;

    for (int c = threadIdx.x; c < CONV; c += 256) {
        float acc = 0.f;
#pragma unroll
        for (int j = 0; j < KW; ++j) {
            int tt = t - (KW - 1) + j;
            if (tt >= 0)
                acc += cw[c * KW + j] * mixed[(size_t)(bt - (KW - 1) + j) * CONV + c];
        }
        yrow[c] = acc / (1.f + expf(-acc)); // SiLU
    }
    __syncthreads();

    const size_t row = (size_t)bt * (H * DK);
    const int g   = threadIdx.x >> 3;   // 0..31 : 16 q-heads then 16 k-heads
    const int sub = threadIdx.x & 7;

    float ss = 0.f;
#pragma unroll
    for (int i = 0; i < 16; ++i) {
        float u = yrow[g * DK + sub * 16 + i];
        ss += u * u;
    }
#pragma unroll
    for (int off = 4; off; off >>= 1) ss += __shfl_xor(ss, off, 8);
    const float sc = rsqrtf(ss + EPSF);

    if (g < 16) {
#pragma unroll
        for (int i = 0; i < 16; ++i) {
            int e = sub * 16 + i;
            qn[row + g * DK + e] = yrow[g * DK + e] * sc;
        }
    } else {
        int h = g - 16;
#pragma unroll
        for (int i = 0; i < 16; ++i) {
            int e = sub * 16 + i;
            kn[row + h * DK + e] = yrow[(16 + h) * DK + e] * sc;
        }
    }
    for (int c = 2 * H * DK + threadIdx.x; c < CONV; c += 256)
        vbuf[row + (c - 2 * H * DK)] = yrow[c];
}

// ---------------------------------------------------------------------------
// Per-head gates.
// ---------------------------------------------------------------------------
__global__ __launch_bounds__(256) void gates_kernel(const float* __restrict__ x,
                                                    const float* __restrict__ wb,
                                                    const float* __restrict__ wa,
                                                    const float* __restrict__ dtb,
                                                    const float* __restrict__ alog,
                                                    float* __restrict__ beta,
                                                    float* __restrict__ decay) {
    __shared__ float xr[Dm];
    const int bt = blockIdx.x;
    for (int i = threadIdx.x; i < Dm; i += 256) xr[i] = x[(size_t)bt * Dm + i];
    __syncthreads();

    const int lane = threadIdx.x & 31;
    const int wv   = threadIdx.x >> 5;
    for (int d = wv; d < 2 * H; d += 8) {
        const int h = d & 15;
        const float* w = (d < H) ? (wb + (size_t)h * Dm) : (wa + (size_t)h * Dm);
        float p = 0.f;
        for (int i = lane; i < Dm; i += 32) p += xr[i] * w[i];
#pragma unroll
        for (int off = 16; off; off >>= 1) p += __shfl_xor(p, off, 32);
        if (lane == 0) {
            if (d < H) {
                beta[bt * H + h] = 1.f / (1.f + expf(-p));
            } else {
                float s  = p + dtb[h];
                float sp = (s > 20.f) ? s : log1pf(expf(s));
                decay[bt * H + h] = expf(-expf(alog[h]) * sp);
            }
        }
    }
}

// ---------------------------------------------------------------------------
// Gated delta-rule recurrence. One block per (b,h); 256 threads = 128 v-cols
// x 2 k-halves; 64 state elements per thread in VGPRs. k/q rows for step t+1
// are prefetched into double-buffered LDS with async-to-LDS DMA while step t
// computes.
// ---------------------------------------------------------------------------
__global__ __launch_bounds__(256) void scan_kernel(const float* __restrict__ qn,
                                                   const float* __restrict__ kn,
                                                   const float* __restrict__ vbuf,
                                                   const float* __restrict__ beta,
                                                   const float* __restrict__ decay,
                                                   float* __restrict__ obuf) {
    __shared__ float ksq[2][256];   // [buf][0:128 k | 128:256 q]
    __shared__ float red1[256], red2[256];
    const int b   = blockIdx.x / H;
    const int h   = blockIdx.x % H;
    const int tid = threadIdx.x;
    const int v   = tid & 127;
    const int kh  = tid >> 7;
    const int base = kh * 64;

    unsigned ldst[2] = { ldsoff(&ksq[0][tid]), ldsoff(&ksq[1][tid]) };
    const float* gsrc0 = (tid < 128) ? kn : qn;
    const int    gcol  = (tid < 128) ? tid : (tid - 128);

    float S[64];
#pragma unroll
    for (int i = 0; i < 64; ++i) S[i] = 0.f;

    // prologue: stage t=0
    {
        size_t row0 = (size_t)(b * Tt) * (H * DV) + (size_t)h * DV;
        async_load_b32(ldst[0], gsrc0 + row0 + gcol);
    }

    for (int t = 0; t < Tt; ++t) {
        const int s = t & 1;
        wait_async0();
        __syncthreads();
        if (t + 1 < Tt) {
            size_t rown = (size_t)(b * Tt + t + 1) * (H * DV) + (size_t)h * DV;
            async_load_b32(ldst[s ^ 1], gsrc0 + rown + gcol);
        }
        const float* ks = &ksq[s][0];
        const float* qs = &ksq[s][128];

        const size_t row = (size_t)(b * Tt + t) * (H * DV) + (size_t)h * DV;
        const float d   = decay[(b * Tt + t) * H + h];
        const float bta = beta[(b * Tt + t) * H + h];
        const float vt  = vbuf[row + v];

        float p = 0.f;
#pragma unroll
        for (int i = 0; i < 64; ++i) p += ks[base + i] * S[i];
        red1[tid] = p;
        __syncthreads();

        const float kv = d * (red1[v] + red1[v + 128]);
        const float u  = (vt - kv) * bta;

        float op = 0.f;
#pragma unroll
        for (int i = 0; i < 64; ++i) {
            S[i] = d * S[i] + ks[base + i] * u;
            op += qs[base + i] * S[i];
        }
        red2[tid] = op;
        __syncthreads();

        if (kh == 0) obuf[row + v] = red2[v] + red2[v + 128];
    }
}

// ---------------------------------------------------------------------------
// Gated RMSNorm per head + silu(z) gate, emit f16 for output GEMM.
// ---------------------------------------------------------------------------
__global__ __launch_bounds__(256) void gated_norm(const float* __restrict__ obuf,
                                                  const float* __restrict__ zbuf,
                                                  const float* __restrict__ nw,
                                                  _Float16* __restrict__ onh) {
    const int bt = blockIdx.x;
    const int h  = threadIdx.x >> 4;
    const int l  = threadIdx.x & 15;
    const size_t row = (size_t)bt * (H * DV) + (size_t)h * DV;

    float ov[8];
    float ss = 0.f;
#pragma unroll
    for (int i = 0; i < 8; ++i) {
        int e = l + 16 * i;
        float o = obuf[row + e];
        ov[i] = o;
        ss += o * o;
    }
#pragma unroll
    for (int off = 8; off; off >>= 1) ss += __shfl_xor(ss, off, 16);
    const float sc = rsqrtf(ss / (float)DV + EPSF);

#pragma unroll
    for (int i = 0; i < 8; ++i) {
        int e = l + 16 * i;
        float z = zbuf[row + e];
        float on = ov[i] * sc * nw[e] * (z / (1.f + expf(-z)));
        onh[row + e] = (_Float16)on;
    }
}

// ---------------------------------------------------------------------------
extern "C" void kernel_launch(void* const* d_in, const int* in_sizes, int n_in,
                              void* d_out, int out_size, void* d_ws, size_t ws_size,
                              hipStream_t stream) {
    const float* x      = (const float*)d_in[0];
    const float* w_qkv  = (const float*)d_in[1];
    const float* conv_w = (const float*)d_in[2];
    const float* w_z    = (const float*)d_in[3];
    const float* w_b    = (const float*)d_in[4];
    const float* w_a    = (const float*)d_in[5];
    const float* dt_b   = (const float*)d_in[6];
    const float* A_log  = (const float*)d_in[7];
    const float* norm_w = (const float*)d_in[8];
    const float* w_out  = (const float*)d_in[9];
    float* out = (float*)d_out;

    char* ws = (char*)d_ws;
    size_t off = 0;
    auto alloc = [&](size_t bytes) -> void* {
        void* p = ws + off;
        off += (bytes + 255) & ~(size_t)255;
        return p;
    };

    _Float16* xh     = (_Float16*)alloc((size_t)BT * Dm * 2);
    _Float16* wqkvh  = (_Float16*)alloc((size_t)CONV * Dm * 2);
    _Float16* wzh    = (_Float16*)alloc((size_t)(H * DV) * Dm * 2);
    _Float16* wouth  = (_Float16*)alloc((size_t)Dm * (H * DV) * 2);
    float*    mixed  = (float*)   alloc((size_t)BT * CONV * 4);
    float*    zbuf   = (float*)   alloc((size_t)BT * (H * DV) * 4);
    float*    qn     = (float*)   alloc((size_t)BT * (H * DK) * 4);
    float*    kn     = (float*)   alloc((size_t)BT * (H * DK) * 4);
    float*    vbuf   = (float*)   alloc((size_t)BT * (H * DV) * 4);
    float*    beta_  = (float*)   alloc((size_t)BT * H * 4);
    float*    decay_ = (float*)   alloc((size_t)BT * H * 4);
    // obuf / onh alias the mixed buffer (mixed is dead after the conv kernel).
    float*    obuf = mixed;
    _Float16* onh  = (_Float16*)((char*)mixed + ((size_t)50 << 20));

    // 1) f32 -> f16 conversions
    {
        int n;
        n = BT * Dm;      cvt_f32_f16<<<(n + 255) / 256, 256, 0, stream>>>(x, xh, n);
        n = CONV * Dm;    cvt_f32_f16<<<(n + 255) / 256, 256, 0, stream>>>(w_qkv, wqkvh, n);
        n = H * DV * Dm;  cvt_f32_f16<<<(n + 255) / 256, 256, 0, stream>>>(w_z, wzh, n);
        n = Dm * H * DV;  cvt_f32_f16<<<(n + 255) / 256, 256, 0, stream>>>(w_out, wouth, n);
    }

    // 2) mixed = x @ w_qkv^T   [4096 x 6144], K = 2048
    gemm_f16_abt<<<dim3(CONV / 128, BT / 128), 256, 0, stream>>>(xh, wqkvh, mixed, BT, CONV, Dm);

    // 3) z = x @ w_z^T         [4096 x 2048]
    gemm_f16_abt<<<dim3((H * DV) / 128, BT / 128), 256, 0, stream>>>(xh, wzh, zbuf, BT, H * DV, Dm);

    // 4) gates (beta, decay)
    gates_kernel<<<BT, 256, 0, stream>>>(x, w_b, w_a, dt_b, A_log, beta_, decay_);

    // 5) causal depthwise conv + SiLU + l2norm(q,k), split v
    conv_silu_norm<<<BT, 256, 0, stream>>>(mixed, conv_w, qn, kn, vbuf);

    // 6) gated delta-rule scan over T (serial), parallel over (b,h)
    scan_kernel<<<Bb * H, 256, 0, stream>>>(qn, kn, vbuf, beta_, decay_, obuf);

    // 7) gated RMSNorm + silu(z) gate -> f16
    gated_norm<<<BT, 256, 0, stream>>>(obuf, zbuf, norm_w, onh);

    // 8) out = on @ w_out^T    [4096 x 2048]
    gemm_f16_abt<<<dim3(Dm / 128, BT / 128), 256, 0, stream>>>(onh, wouth, out, BT, Dm, H * DV);
}
